// CausalSelfAttentionLayer_58093727646313
// MI455X (gfx1250) — matrix operand order repaired and checked by reading, not compiled, not run on hardware
//
#include <hip/hip_runtime.h>

typedef __attribute__((ext_vector_type(16))) _Float16 v16h;
typedef __attribute__((ext_vector_type(8)))  _Float16 v8h;
typedef __attribute__((ext_vector_type(8)))  float    v8f;

#define Hdim 2048
#define Ddim 128
#define NQH  16
#define NKVH 16
#define Bsz  4
#define Lq   1024
#define Cc   512
#define Sfull 1536            // C + L
#define Ttot (Bsz * Lq)       // 4096
#define QKVN (Ddim * (NQH + 2 * NKVH))  // 6144

// ---------------------------------------------------------------------------
// WMMA fragment loaders (contiguous k order used for both A and B, f16, 16x16x32)
// A (16x32, MxK): lane l -> row (l&15); lanes 0-15 hold K=0..15 contiguous,
//                 lanes 16-31 hold K=16..31 contiguous     -> one b256 load
// B (32x16, KxN): lane l -> col (l&15); lanes 0-15 hold K=0..15 contiguous,
//                 lanes 16-31 hold K=16..31 contiguous     -> one b256 load
// C/D (16x16 f32): elem r -> row r + hi*8, col (lane&15)
// ---------------------------------------------------------------------------
__device__ __forceinline__ v16h load_frag_a(const _Float16* base, size_t stride, int lane) {
  const int r  = lane & 15;
  const int hi = (lane >> 4) & 1;
  const _Float16* p = base + (size_t)r * stride + hi * 16;
  return *reinterpret_cast<const v16h*>(p);
}

__device__ __forceinline__ v16h load_frag_b(const _Float16* base, size_t stride, int lane) {
  const int r  = lane & 15;
  const int hi = (lane >> 4) & 1;
  return *reinterpret_cast<const v16h*>(base + (size_t)r * stride + hi * 16);
}

__device__ __forceinline__ float redmax16(float v) {
#pragma unroll
  for (int o = 1; o < 16; o <<= 1) v = fmaxf(v, __shfl_xor(v, o, 16));
  return v;
}
__device__ __forceinline__ float redsum16(float v) {
#pragma unroll
  for (int o = 1; o < 16; o <<= 1) v += __shfl_xor(v, o, 16);
  return v;
}

// ---------------------------------------------------------------------------
// 1) Transpose + fp32->f16 convert:  in[R][C] -> out[C][R]
// ---------------------------------------------------------------------------
__global__ __launch_bounds__(256) void transpose_f32_f16(
    const float* __restrict__ in, _Float16* __restrict__ out, int R, int C) {
  __shared__ float tile[32][33];
  const int c0 = blockIdx.x * 32;
  const int r0 = blockIdx.y * 32;
  const int tx = threadIdx.x, ty = threadIdx.y;   // block (32,8)
#pragma unroll
  for (int i = 0; i < 4; ++i) {
    int r = r0 + ty + i * 8;
    tile[ty + i * 8][tx] = in[(size_t)r * C + c0 + tx];
  }
  __syncthreads();
#pragma unroll
  for (int i = 0; i < 4; ++i) {
    int c = c0 + ty + i * 8;
    out[(size_t)c * R + r0 + tx] = (_Float16)tile[tx][ty + i * 8];
  }
}

// ---------------------------------------------------------------------------
// 2) LayerNorm -> f16
// ---------------------------------------------------------------------------
__global__ __launch_bounds__(256) void layernorm_f16(
    const float* __restrict__ x, const float* __restrict__ w,
    const float* __restrict__ b, _Float16* __restrict__ out) {
  const int t = blockIdx.x;
  const float* row = x + (size_t)t * Hdim;
  float lsum = 0.f, lsq = 0.f;
  for (int i = threadIdx.x; i < Hdim; i += 256) {
    float v = row[i];
    lsum += v; lsq += v * v;
  }
  __shared__ float s1[256], s2[256];
  s1[threadIdx.x] = lsum; s2[threadIdx.x] = lsq;
  __syncthreads();
  for (int o = 128; o > 0; o >>= 1) {
    if (threadIdx.x < o) { s1[threadIdx.x] += s1[threadIdx.x + o];
                           s2[threadIdx.x] += s2[threadIdx.x + o]; }
    __syncthreads();
  }
  const float mu   = s1[0] * (1.f / Hdim);
  const float var  = s2[0] * (1.f / Hdim) - mu * mu;
  const float rstd = rsqrtf(var + 1e-5f);
  for (int i = threadIdx.x; i < Hdim; i += 256) {
    float v = (row[i] - mu) * rstd * w[i] + b[i];
    out[(size_t)t * Hdim + i] = (_Float16)v;
  }
}

// ---------------------------------------------------------------------------
// 3) Cache conversion
//    k_cache[b][s][h][d] (fp32, s<512) -> k_full[b*16+h][s][d] (f16)
//    v_cache -> vT[b*16+h][d][s] (f16)
// ---------------------------------------------------------------------------
__global__ __launch_bounds__(256) void kcache_convert(
    const float* __restrict__ kc, _Float16* __restrict__ kf) {
  int idx = blockIdx.x * 256 + threadIdx.x;          // B*C*NKV*D = 4194304
  if (idx >= Bsz * Cc * NKVH * Ddim) return;
  int d = idx & 127; int r = idx >> 7;
  int h = r & 15;    r >>= 4;
  int s = r & 511;   int b = r >> 9;
  float v = kc[((((size_t)b * 2048 + s) * NKVH + h) * Ddim) + d];
  kf[(((size_t)(b * NKVH + h)) * Sfull + s) * Ddim + d] = (_Float16)v;
}

__global__ __launch_bounds__(256) void vcache_convert(
    const float* __restrict__ vc, _Float16* __restrict__ vT) {
  int idx = blockIdx.x * 256 + threadIdx.x;
  if (idx >= Bsz * Cc * NKVH * Ddim) return;
  int d = idx & 127; int r = idx >> 7;
  int h = r & 15;    r >>= 4;
  int s = r & 511;   int b = r >> 9;
  float v = vc[((((size_t)b * 2048 + s) * NKVH + h) * Ddim) + d];
  vT[((((size_t)(b * NKVH + h)) * Ddim + d) * Sfull) + s] = (_Float16)v;
}

// ---------------------------------------------------------------------------
// 4) WMMA GEMM: C[M,N] = A[M,K](f16, row-major) * W (given as BT[N][K] f16)
//    one wave per 64x128 output block: 4 m-subtiles x 8 n-subtiles,
//    32 v8f accumulators (256 VGPRs), A-frags reused 8x, B-frags reused 4x.
// ---------------------------------------------------------------------------
template <bool OUT_F32>
__global__ __launch_bounds__(32) void gemm_wmma_f16(
    const _Float16* __restrict__ A, const _Float16* __restrict__ BT,
    const float* __restrict__ bias, _Float16* __restrict__ Cf16,
    float* __restrict__ Cf32, int M, int N, int K) {
  const int lane = threadIdx.x;
  const int r15 = lane & 15, hi = lane >> 4;
  const int m0 = blockIdx.x * 64;
  const int n0 = blockIdx.y * 128;
  v8f acc[4][8] = {};
  for (int k = 0; k < K; k += 32) {
    v16h af[4];
#pragma unroll
    for (int i = 0; i < 4; ++i)
      af[i] = load_frag_a(A + (size_t)(m0 + i * 16) * K + k, K, lane);
    // prefetch next A K-chunk (lowers to global_prefetch_b8)
    if (k + 32 < K)
      __builtin_prefetch(A + (size_t)(m0 + r15) * K + k + 32, 0, 0);
#pragma unroll
    for (int j = 0; j < 8; ++j) {
      v16h bf = load_frag_b(BT + (size_t)(n0 + j * 16) * K + k, K, lane);
#pragma unroll
      for (int i = 0; i < 4; ++i)
        acc[i][j] = __builtin_amdgcn_wmma_f32_16x16x32_f16(
            false, af[i], false, bf, (short)0, acc[i][j], false, false);
    }
  }
#pragma unroll
  for (int i = 0; i < 4; ++i) {
#pragma unroll
    for (int j = 0; j < 8; ++j) {
#pragma unroll
      for (int r = 0; r < 8; ++r) {
        int m = m0 + i * 16 + r + hi * 8;
        int c = n0 + j * 16 + r15;
        float v = acc[i][j][r] + bias[c];
        if (OUT_F32) Cf32[(size_t)m * N + c] = v;
        else         Cf16[(size_t)m * N + c] = (_Float16)v;
      }
    }
  }
}

// ---------------------------------------------------------------------------
// 5) Scatter qkv[T][6144] -> q[bh][l][d], k_full[bh][512+l][d], vT[bh][d][512+l]
// ---------------------------------------------------------------------------
__global__ __launch_bounds__(256) void scatter_qkv(
    const _Float16* __restrict__ qkv, _Float16* __restrict__ q,
    _Float16* __restrict__ kf, _Float16* __restrict__ vT) {
  size_t idx = (size_t)blockIdx.x * 256 + threadIdx.x;
  if (idx >= (size_t)Ttot * QKVN) return;
  int c = (int)(idx % QKVN);
  size_t t = idx / QKVN;
  int b  = (int)(t >> 10);
  int li = (int)(t & 1023);
  _Float16 v = qkv[idx];
  if (c < Ddim * NQH) {
    int h = c >> 7, d = c & 127;
    q[(((size_t)(b * NQH + h)) * Lq + li) * Ddim + d] = v;
  } else if (c < Ddim * (NQH + NKVH)) {
    int cc = c - Ddim * NQH; int h = cc >> 7, d = cc & 127;
    kf[(((size_t)(b * NKVH + h)) * Sfull + Cc + li) * Ddim + d] = v;
  } else {
    int cc = c - Ddim * (NQH + NKVH); int h = cc >> 7, d = cc & 127;
    vT[((((size_t)(b * NKVH + h)) * Ddim + d) * Sfull) + Cc + li] = v;
  }
}

// ---------------------------------------------------------------------------
// 6) Flash attention: one wave per (qtile=16 rows, head, batch)
//    scores: A=Q(16xD), B=K^T via k_full rows; PV: A=P(16x32), B=vT rows
// ---------------------------------------------------------------------------
__global__ __launch_bounds__(32) void attn_flash(
    const _Float16* __restrict__ q, const _Float16* __restrict__ kf,
    const _Float16* __restrict__ vT, _Float16* __restrict__ ao) {
  const int lane = threadIdx.x;
  const int r15 = lane & 15, hi = lane >> 4;
  const int qt = blockIdx.x, h = blockIdx.y, b = blockIdx.z;
  const int qbase = qt * 16;
  const int bh = b * NKVH + h;
  const _Float16* qtile = q  + (((size_t)bh * Lq + qbase) * Ddim);
  const _Float16* ktile = kf + ((size_t)bh * Sfull * Ddim);
  const _Float16* vtile = vT + ((size_t)bh * Ddim * Sfull);
  const float scale = 0.08838834764831845f;   // 1/sqrt(128)

  v16h qf[4];
#pragma unroll
  for (int d = 0; d < 4; ++d) qf[d] = load_frag_a(qtile + d * 32, Ddim, lane);

  v8f out[8] = {};
  float Mrow[8], Lrow[8];
#pragma unroll
  for (int r = 0; r < 8; ++r) { Mrow[r] = -3.0e38f; Lrow[r] = 0.f; }

  __shared__ __align__(32) _Float16 lds_p[16 * 32];
  const int kmax = Cc + qbase + 16;

  for (int kb = 0; kb < kmax; kb += 32) {
    v8f s0 = {}, s1 = {};
#pragma unroll
    for (int d = 0; d < 4; ++d) {
      v16h k0 = load_frag_b(ktile + (size_t)kb * Ddim + d * 32, Ddim, lane);
      v16h k1 = load_frag_b(ktile + (size_t)(kb + 16) * Ddim + d * 32, Ddim, lane);
      s0 = __builtin_amdgcn_wmma_f32_16x16x32_f16(false, qf[d], false, k0, (short)0, s0, false, false);
      s1 = __builtin_amdgcn_wmma_f32_16x16x32_f16(false, qf[d], false, k1, (short)0, s1, false, false);
    }
    float p0[8], p1[8];
#pragma unroll
    for (int r = 0; r < 8; ++r) {
      const int pos  = Cc + qbase + r + hi * 8;
      const int key0 = kb + r15;
      const int key1 = kb + 16 + r15;
      float a0 = (key0 <= pos) ? s0[r] : -1.0e30f;
      float a1 = (key1 <= pos) ? s1[r] : -1.0e30f;
      float t  = redmax16(fmaxf(a0, a1));
      float mn = fmaxf(Mrow[r], t);
      float al = __expf((Mrow[r] - mn) * scale);
      Mrow[r] = mn;
      p0[r] = __expf((a0 - mn) * scale);
      p1[r] = __expf((a1 - mn) * scale);
      float rs = redsum16(p0[r] + p1[r]);
      Lrow[r] = Lrow[r] * al + rs;
#pragma unroll
      for (int j = 0; j < 8; ++j) out[j][r] *= al;
    }
    // transpose P (C-layout) -> A-layout via LDS
#pragma unroll
    for (int r = 0; r < 8; ++r) {
      int m = r + hi * 8;
      lds_p[m * 32 + r15]      = (_Float16)p0[r];
      lds_p[m * 32 + 16 + r15] = (_Float16)p1[r];
    }
    __syncthreads();
    v16h pf = load_frag_a(lds_p, 32, lane);
    __syncthreads();
#pragma unroll
    for (int j = 0; j < 8; ++j) {
      v16h vf = load_frag_b(vtile + (size_t)(j * 16) * Sfull + kb, Sfull, lane);
      out[j] = __builtin_amdgcn_wmma_f32_16x16x32_f16(false, pf, false, vf, (short)0, out[j], false, false);
    }
  }
  const size_t trow = (size_t)b * Lq + qbase;
#pragma unroll
  for (int j = 0; j < 8; ++j) {
#pragma unroll
    for (int r = 0; r < 8; ++r) {
      int m = r + hi * 8;
      float v = out[j][r] / Lrow[r];
      ao[(trow + m) * Hdim + h * Ddim + j * 16 + r15] = (_Float16)v;
    }
  }
}

// ---------------------------------------------------------------------------
// launch
// ---------------------------------------------------------------------------
extern "C" void kernel_launch(void* const* d_in, const int* in_sizes, int n_in,
                              void* d_out, int out_size, void* d_ws, size_t ws_size,
                              hipStream_t stream) {
  const float* hidden  = (const float*)d_in[0];
  const float* k_cache = (const float*)d_in[2];
  const float* v_cache = (const float*)d_in[3];
  const float* ln_w    = (const float*)d_in[6];
  const float* ln_b    = (const float*)d_in[7];
  const float* attn_w  = (const float*)d_in[8];
  const float* attn_b  = (const float*)d_in[9];
  const float* proj_w  = (const float*)d_in[10];
  const float* proj_b  = (const float*)d_in[11];
  float* out = (float*)d_out;

  char* ws = (char*)d_ws;
  size_t off = 0;
  auto alloc = [&](size_t bytes) { char* p = ws + off; off = (off + bytes + 255) & ~(size_t)255; return p; };
  _Float16* xln   = (_Float16*)alloc((size_t)Ttot * Hdim * 2);        // 16 MB
  _Float16* awT   = (_Float16*)alloc((size_t)QKVN * Hdim * 2);        // 24 MB
  _Float16* pwT   = (_Float16*)alloc((size_t)Hdim * Hdim * 2);        //  8 MB
  _Float16* qkv16 = (_Float16*)alloc((size_t)Ttot * QKVN * 2);        // 48 MB
  _Float16* q16   = (_Float16*)alloc((size_t)Bsz * NQH * Lq * Ddim * 2);
  _Float16* kful  = (_Float16*)alloc((size_t)Bsz * NKVH * Sfull * Ddim * 2);
  _Float16* vT16  = (_Float16*)alloc((size_t)Bsz * NKVH * Ddim * Sfull * 2);
  _Float16* ao16  = (_Float16*)alloc((size_t)Ttot * Hdim * 2);

  // weight transpose+convert: attn_w [H][QKVN] -> awT [QKVN][H]
  transpose_f32_f16<<<dim3(QKVN / 32, Hdim / 32), dim3(32, 8), 0, stream>>>(attn_w, awT, Hdim, QKVN);
  transpose_f32_f16<<<dim3(Hdim / 32, Hdim / 32), dim3(32, 8), 0, stream>>>(proj_w, pwT, Hdim, Hdim);

  // layernorm
  layernorm_f16<<<Ttot, 256, 0, stream>>>(hidden, ln_w, ln_b, xln);

  // cache conversion
  {
    int n = Bsz * Cc * NKVH * Ddim;
    kcache_convert<<<(n + 255) / 256, 256, 0, stream>>>(k_cache, kful);
    vcache_convert<<<(n + 255) / 256, 256, 0, stream>>>(v_cache, vT16);
  }

  // QKV GEMM: [4096,2048] x [2048,6144], 64x128 per wave
  gemm_wmma_f16<false><<<dim3(Ttot / 64, QKVN / 128), 32, 0, stream>>>(
      xln, awT, attn_b, qkv16, nullptr, Ttot, QKVN, Hdim);

  // scatter q / k / v
  {
    size_t n = (size_t)Ttot * QKVN;
    scatter_qkv<<<(unsigned)((n + 255) / 256), 256, 0, stream>>>(qkv16, q16, kful, vT16);
  }

  // flash attention
  attn_flash<<<dim3(Lq / 16, NQH, Bsz), 32, 0, stream>>>(q16, kful, vT16, ao16);

  // output projection -> fp32 d_out, 64x128 per wave
  gemm_wmma_f16<true><<<dim3(Ttot / 64, Hdim / 128), 32, 0, stream>>>(
      ao16, pwT, proj_b, nullptr, out, Ttot, Hdim, Hdim);
}
